// GraphModule_10720238370828
// MI455X (gfx1250) — matrix operand order repaired and checked by reading, compiled, not run on hardware
//
#include <hip/hip_runtime.h>

#define NB   8
#define NN   2048
#define ND   128
#define NOUT 128

typedef __attribute__((ext_vector_type(16))) __bf16 v16bf;
typedef __attribute__((ext_vector_type(8)))  float  v8f;

union Frag { v16bf v; uint4 q[2]; unsigned short u[16]; };
union Acc  { v8f v; float f[8]; };

// ---------- f32 -> bf16 split (round-to-nearest-even, hi + lo) ----------
static __device__ __forceinline__ unsigned short bf16_rne(float f) {
  unsigned int u = __float_as_uint(f);
  u += 0x7FFFu + ((u >> 16) & 1u);
  return (unsigned short)(u >> 16);
}
static __device__ __forceinline__ float bf16_f32(unsigned short h) {
  return __uint_as_float(((unsigned int)h) << 16);
}
static __device__ __forceinline__ void split_f32(float x, unsigned short& hi, unsigned short& lo) {
  hi = bf16_rne(x);
  lo = bf16_rne(x - bf16_f32(hi));
}

static __device__ __forceinline__ v16bf ldfrag(const unsigned short* p) {
  Frag f;
  const uint4* q = (const uint4*)p;
  f.q[0] = q[0]; f.q[1] = q[1];
  return f.v;
}
static __device__ __forceinline__ v8f wmma_bf16(v16bf a, v16bf b, v8f c) {
  // D = A(16x32 bf16) x B(32x16 bf16) + C(16x16 f32)
  return __builtin_amdgcn_wmma_f32_16x16x32_bf16(false, a, false, b, (short)0, c, false, false);
}

// ---------- prep: per-row dots sj = x.w_j, si = x.w_i ----------
__global__ __launch_bounds__(256) void k_sisj(const float* __restrict__ inp,
                                              const float* __restrict__ g_w,
                                              float* __restrict__ si, float* __restrict__ sj) {
  int tid = blockIdx.x * 256 + threadIdx.x;
  if (tid >= NB * NN) return;
  const float4* row = (const float4*)(inp + (size_t)tid * ND);
  const float4* wj  = (const float4*)(g_w);
  const float4* wi  = (const float4*)(g_w + ND);
  float aj = 0.f, ai = 0.f;
#pragma unroll
  for (int q = 0; q < ND / 4; ++q) {
    float4 x = row[q]; float4 a = wj[q]; float4 b = wi[q];
    aj = fmaf(x.x, a.x, fmaf(x.y, a.y, fmaf(x.z, a.z, fmaf(x.w, a.w, aj))));
    ai = fmaf(x.x, b.x, fmaf(x.y, b.y, fmaf(x.z, b.z, fmaf(x.w, b.w, ai))));
  }
  sj[tid] = aj; si[tid] = ai;
}

// ---------- prep: relation -> A-fragment layout (bf16 hi/lo) ----------
// A 16x32 bf16 layout: lane l holds row M=l&15; elems e=0..15 -> K=(l>>4)*16+e
__global__ __launch_bounds__(256) void k_relfrag(const float* __restrict__ rel,
                                                 unsigned short* __restrict__ fh,
                                                 unsigned short* __restrict__ fl) {
  int tid  = blockIdx.x * 256 + threadIdx.x;  // ((rt*64 + kc)*32 + lane), 128*64*32 total
  int lane = tid & 31;
  int kc   = (tid >> 5) & 63;
  int rt   = tid >> 11;
  int h = lane >> 4, ln = lane & 15;
  const float* r = rel + (size_t)(rt * 16 + ln) * NN + kc * 32;
  Frag H, L;
#pragma unroll
  for (int j = 0; j < 8; ++j) {
    split_f32(r[h * 16 + j],     H.u[j],     L.u[j]);
    split_f32(r[h * 16 + 8 + j], H.u[8 + j], L.u[8 + j]);
  }
  uint4* dh = (uint4*)fh + (size_t)tid * 2;
  uint4* dl = (uint4*)fl + (size_t)tid * 2;
  dh[0] = H.q[0]; dh[1] = H.q[1];
  dl[0] = L.q[0]; dl[1] = L.q[1];
}

// ---------- prep: inputs -> B-fragment layout (bf16 hi/lo) ----------
// B 32x16 bf16 layout: lane l holds col N=l&15; elems e=0..15 -> K=(l>>4)*16+e
__global__ __launch_bounds__(256) void k_infrag(const float* __restrict__ inp,
                                                unsigned short* __restrict__ fh,
                                                unsigned short* __restrict__ fl) {
  int tid  = blockIdx.x * 256 + threadIdx.x;  // (((b*64+kc)*8+t)*32+lane), 8*64*8*32 total
  int lane = tid & 31;
  int t    = (tid >> 5) & 7;
  int kc   = (tid >> 8) & 63;
  int b    = tid >> 14;
  int h = lane >> 4, ln = lane & 15;
  const float* p = inp + ((size_t)b * NN + kc * 32 + h * 16) * ND + t * 16 + ln;
  Frag H, L;
#pragma unroll
  for (int e = 0; e < 16; ++e)
    split_f32(p[(size_t)e * ND], H.u[e], L.u[e]);
  uint4* dh = (uint4*)fh + (size_t)tid * 2;
  uint4* dl = (uint4*)fl + (size_t)tid * 2;
  dh[0] = H.q[0]; dh[1] = H.q[1];
  dl[0] = L.q[0]; dl[1] = L.q[1];
}

// ---------- prep: weight -> B-fragment layout (bf16 hi/lo), K=D=128 ----------
__global__ __launch_bounds__(256) void k_wfrag(const float* __restrict__ w,
                                               unsigned short* __restrict__ fh,
                                               unsigned short* __restrict__ fl) {
  int tid = blockIdx.x * 256 + threadIdx.x;  // ((kc*8+t)*32+lane), 4*8*32 total
  if (tid >= 1024) return;
  int lane = tid & 31;
  int t    = (tid >> 5) & 7;
  int kc   = tid >> 8;
  int h = lane >> 4, ln = lane & 15;
  const float* p = w + (size_t)(kc * 32 + h * 16) * NOUT + t * 16 + ln;
  Frag H, L;
#pragma unroll
  for (int e = 0; e < 16; ++e)
    split_f32(p[(size_t)e * NOUT], H.u[e], L.u[e]);
  uint4* dh = (uint4*)fh + (size_t)tid * 2;
  uint4* dl = (uint4*)fl + (size_t)tid * 2;
  dh[0] = H.q[0]; dh[1] = H.q[1];
  dl[0] = L.q[0]; dl[1] = L.q[1];
}

// ---------- prep: S = sum_n x, T = sum_n sj*x (two-stage, deterministic) ----------
__global__ __launch_bounds__(128) void k_st_part(const float* __restrict__ inp,
                                                 const float* __restrict__ sj,
                                                 float* __restrict__ Sp, float* __restrict__ Tp) {
  int c = blockIdx.x, b = blockIdx.y, d = threadIdx.x;
  const float* base = inp + ((size_t)b * NN + c * 128) * ND + d;
  const float* sjb  = sj + b * NN + c * 128;
  float s = 0.f, tt = 0.f;
#pragma unroll 4
  for (int n = 0; n < 128; ++n) {
    float x = base[(size_t)n * ND];
    s += x;
    tt = fmaf(sjb[n], x, tt);
  }
  Sp[(b * 16 + c) * ND + d] = s;
  Tp[(b * 16 + c) * ND + d] = tt;
}

__global__ __launch_bounds__(128) void k_st_red(const float* __restrict__ Sp,
                                                const float* __restrict__ Tp,
                                                float* __restrict__ S, float* __restrict__ T) {
  int tid = blockIdx.x * 128 + threadIdx.x;  // b*128 + d
  if (tid >= NB * ND) return;
  int b = tid >> 7, d = tid & 127;
  float s = 0.f, t = 0.f;
#pragma unroll
  for (int c = 0; c < 16; ++c) {
    s += Sp[(b * 16 + c) * ND + d];
    t += Tp[(b * 16 + c) * ND + d];
  }
  S[tid] = s; T[tid] = t;
}

// ---------- main: R = rel @ x[b] via split-bf16 WMMA, fused epilogue + @weight ----------
__global__ __launch_bounds__(128) void k_main(
    const unsigned short* __restrict__ relf_h, const unsigned short* __restrict__ relf_l,
    const unsigned short* __restrict__ inf_h,  const unsigned short* __restrict__ inf_l,
    const unsigned short* __restrict__ wf_h,   const unsigned short* __restrict__ wf_l,
    const float* __restrict__ si, const float* __restrict__ Sv, const float* __restrict__ Tv,
    const float* __restrict__ g_w, const float* __restrict__ g_b,
    const float* __restrict__ bias, float* __restrict__ out) {
  __shared__ float lds_agg[4][16][ND + 4];  // +4 pad: spread LDS banks on row reads

  const int b    = blockIdx.y;
  const int wave = threadIdx.x >> 5;
  const int lane = threadIdx.x & 31;
  const int h = lane >> 4, ln = lane & 15;
  const int rt = blockIdx.x * 4 + wave;  // 16-row tile index (0..127)
  const int i0 = rt * 16;

  const float w_r = g_w[2 * ND];
  const float gb  = g_b[0];

  float Sreg[8], Treg[8], Breg[8], sgb[8];
#pragma unroll
  for (int t = 0; t < 8; ++t) {
    Sreg[t] = Sv[b * ND + t * 16 + ln];
    Treg[t] = Tv[b * ND + t * 16 + ln];
    Breg[t] = bias[t * 16 + ln];
  }
#pragma unroll
  for (int r = 0; r < 8; ++r)
    sgb[r] = si[b * NN + i0 + r + 8 * h] + gb;

  Acc acc[8];
#pragma unroll
  for (int t = 0; t < 8; ++t)
#pragma unroll
    for (int r = 0; r < 8; ++r) acc[t].f[r] = 0.f;

  const unsigned short* ah  = relf_h + ((size_t)rt * 64) * 512 + lane * 16;
  const unsigned short* al  = relf_l + ((size_t)rt * 64) * 512 + lane * 16;
  const unsigned short* bh0 = inf_h + ((size_t)b * 64 * 8) * 512 + lane * 16;
  const unsigned short* bl0 = inf_l + ((size_t)b * 64 * 8) * 512 + lane * 16;

  for (int kc = 0; kc < 64; ++kc) {
    v16bf a_h = ldfrag(ah + (size_t)kc * 512);
    v16bf a_l = ldfrag(al + (size_t)kc * 512);
    const unsigned short* bh = bh0 + (size_t)kc * 8 * 512;
    const unsigned short* bl = bl0 + (size_t)kc * 8 * 512;
#pragma unroll
    for (int t = 0; t < 8; ++t) {
      v16bf b_h = ldfrag(bh + (size_t)t * 512);
      v16bf b_l = ldfrag(bl + (size_t)t * 512);
      acc[t].v = wmma_bf16(a_h, b_h, acc[t].v);
      acc[t].v = wmma_bf16(a_h, b_l, acc[t].v);
      acc[t].v = wmma_bf16(a_l, b_h, acc[t].v);
    }
  }

  // agg = (si+g_b)*S + T + w_r*R   -> LDS (re-layout C -> A fragment via LDS)
#pragma unroll
  for (int t = 0; t < 8; ++t)
#pragma unroll
    for (int r = 0; r < 8; ++r) {
      float aggv = fmaf(w_r, acc[t].f[r], fmaf(sgb[r], Sreg[t], Treg[t]));
      lds_agg[wave][r + 8 * h][t * 16 + ln] = aggv;
    }
  __syncthreads();

  Acc oacc[8];
#pragma unroll
  for (int t = 0; t < 8; ++t)
#pragma unroll
    for (int r = 0; r < 8; ++r) oacc[t].f[r] = 0.f;

#pragma unroll
  for (int kc = 0; kc < 4; ++kc) {  // K = D = 128
    Frag AH, AL;
#pragma unroll
    for (int j = 0; j < 8; ++j) {
      split_f32(lds_agg[wave][ln][kc * 32 + h * 16 + j],     AH.u[j],     AL.u[j]);
      split_f32(lds_agg[wave][ln][kc * 32 + h * 16 + 8 + j], AH.u[8 + j], AL.u[8 + j]);
    }
    const unsigned short* wh = wf_h + ((size_t)kc * 8) * 512 + lane * 16;
    const unsigned short* wl = wf_l + ((size_t)kc * 8) * 512 + lane * 16;
#pragma unroll
    for (int t = 0; t < 8; ++t) {
      v16bf w_hf = ldfrag(wh + (size_t)t * 512);
      v16bf w_lf = ldfrag(wl + (size_t)t * 512);
      oacc[t].v = wmma_bf16(AH.v, w_hf, oacc[t].v);
      oacc[t].v = wmma_bf16(AH.v, w_lf, oacc[t].v);
      oacc[t].v = wmma_bf16(AL.v, w_hf, oacc[t].v);
    }
  }

#pragma unroll
  for (int t = 0; t < 8; ++t)
#pragma unroll
    for (int r = 0; r < 8; ++r)
      out[((size_t)b * NN + i0 + r + 8 * h) * NOUT + t * 16 + ln] = oacc[t].f[r] + Breg[t];
}

extern "C" void kernel_launch(void* const* d_in, const int* in_sizes, int n_in,
                              void* d_out, int out_size, void* d_ws, size_t ws_size,
                              hipStream_t stream) {
  (void)in_sizes; (void)n_in; (void)out_size; (void)ws_size;
  const float* inp  = (const float*)d_in[0];
  const float* rel  = (const float*)d_in[1];
  const float* g_w  = (const float*)d_in[2];
  const float* g_b  = (const float*)d_in[3];
  const float* wgt  = (const float*)d_in[4];
  const float* bias = (const float*)d_in[5];
  float* out = (float*)d_out;

  char* ws = (char*)d_ws;
  size_t off = 0;
  auto carve = [&](size_t bytes) -> char* {
    char* p = ws + off;
    off += (bytes + 255) & ~(size_t)255;
    return p;
  };

  unsigned short* relf_h = (unsigned short*)carve((size_t)128 * 64 * 32 * 16 * 2);  // 8.39 MB
  unsigned short* relf_l = (unsigned short*)carve((size_t)128 * 64 * 32 * 16 * 2);
  unsigned short* inf_h  = (unsigned short*)carve((size_t)NB * 64 * 8 * 32 * 16 * 2);  // 4.19 MB
  unsigned short* inf_l  = (unsigned short*)carve((size_t)NB * 64 * 8 * 32 * 16 * 2);
  unsigned short* wf_h   = (unsigned short*)carve((size_t)4 * 8 * 32 * 16 * 2);  // 32 KB
  unsigned short* wf_l   = (unsigned short*)carve((size_t)4 * 8 * 32 * 16 * 2);
  float* si = (float*)carve((size_t)NB * NN * 4);
  float* sj = (float*)carve((size_t)NB * NN * 4);
  float* S  = (float*)carve((size_t)NB * ND * 4);
  float* T  = (float*)carve((size_t)NB * ND * 4);
  float* Sp = (float*)carve((size_t)NB * 16 * ND * 4);
  float* Tp = (float*)carve((size_t)NB * 16 * ND * 4);

  k_sisj   <<<64,          256, 0, stream>>>(inp, g_w, si, sj);
  k_relfrag<<<1024,        256, 0, stream>>>(rel, relf_h, relf_l);
  k_infrag <<<512,         256, 0, stream>>>(inp, inf_h, inf_l);
  k_wfrag  <<<4,           256, 0, stream>>>(wgt, wf_h, wf_l);
  k_st_part<<<dim3(16, 8), 128, 0, stream>>>(inp, sj, Sp, Tp);
  k_st_red <<<8,           128, 0, stream>>>(Sp, Tp, S, T);
  k_main   <<<dim3(32, 8), 128, 0, stream>>>(relf_h, relf_l, inf_h, inf_l, wf_h, wf_l,
                                             si, S, T, g_w, g_b, bias, out);
}
